// DotProjectionAttention_39857296507633
// MI455X (gfx1250) — compile-verified, matrix operand order checked by reading, not run on hardware
//
#include <hip/hip_runtime.h>
#include <hip/hip_bf16.h>

typedef __attribute__((ext_vector_type(16))) __bf16 v16bf;
typedef __attribute__((ext_vector_type(8)))  __bf16 v8bf;
typedef __attribute__((ext_vector_type(4)))  __bf16 v4bf;
typedef __attribute__((ext_vector_type(8)))  float  v8f;

#define B_    64
#define S_    2048
#define ENC_  1024
#define DEC_  1024
#define ATTN_ 1024
#define KDIM  1024

// workspace layout (bytes)
#define WEB_OFF   0                            // bf16 We (ATTN x ENC), 2 MB
#define HPROJ_OFF (ATTN_ * ENC_ * 2)           // f32 hproj (B x ATTN), 256 KB
#define SCORE_OFF (HPROJ_OFF + B_ * ATTN_ * 4) // f32 scores (S*B), 512 KB

// ---------------------------------------------------------------------------
// Kernel 1: convert We = attn_w_weight[:, :ENC] to bf16, row-major (a, e)
// ---------------------------------------------------------------------------
__global__ __launch_bounds__(256) void prep_web_kernel(
    const float* __restrict__ W, __bf16* __restrict__ Web) {
  int i = (blockIdx.x * blockDim.x + threadIdx.x) * 4;  // over ATTN_*ENC_
  int a = i / ENC_;
  int e = i % ENC_;
  float4 w4 = *(const float4*)(W + (size_t)a * (ENC_ + DEC_) + e);
  v4bf o;
  o[0] = (__bf16)w4.x; o[1] = (__bf16)w4.y;
  o[2] = (__bf16)w4.z; o[3] = (__bf16)w4.w;
  *(v4bf*)(Web + (size_t)a * ENC_ + e) = o;
}

// ---------------------------------------------------------------------------
// Kernel 2: hproj[b][a] = sum_d hidden[b,d] * W[a, ENC+d]  + w_bias[a]
// ---------------------------------------------------------------------------
__global__ __launch_bounds__(256) void prep_hproj_kernel(
    const float* __restrict__ hidden, const float* __restrict__ W,
    const float* __restrict__ wb, float* __restrict__ hproj) {
  int i = blockIdx.x * blockDim.x + threadIdx.x;  // over B_*ATTN_
  int b = i / ATTN_;
  int a = i % ATTN_;
  const float4* h4 = (const float4*)(hidden + (size_t)b * DEC_);
  const float4* w4 = (const float4*)(W + (size_t)a * (ENC_ + DEC_) + ENC_);
  float acc = 0.f;
#pragma unroll 4
  for (int d = 0; d < DEC_ / 4; ++d) {
    float4 hv = h4[d], wv = w4[d];
    acc += hv.x * wv.x + hv.y * wv.y + hv.z * wv.z + hv.w * wv.w;
  }
  hproj[i] = acc + wb[a];
}

// ---------------------------------------------------------------------------
// Kernel 3: fused  scores[r] = sum_a vw[a]*tanh( v[r,:]·We[a,:] + hproj[b,a] )
//   r = s*B + b  (matches the natural (S,B,ENC) layout of vectors_to_attend)
//   workgroup: 32 rows x 1024 cols, 8 waves in 2(M) x 4(N) grid,
//   each wave: 16 rows x 256 cols = 16 WMMA accumulators, K looped by 32.
//   LDS A-tile is double-buffered: one barrier per K-step, global A load
//   overlapped with the 16 WMMAs of the current step.
// ---------------------------------------------------------------------------
__device__ __forceinline__ float fast_tanh(float x) {
  x = fminf(fmaxf(x, -15.f), 15.f);
  float e = __expf(2.f * x);
  return (e - 1.f) / (e + 1.f);
}

__global__ __launch_bounds__(256) void attn_scores_kernel(
    const float* __restrict__ v,      // (S*B, ENC) f32
    const __bf16* __restrict__ Web,   // (ATTN, ENC) bf16, L2-resident
    const float* __restrict__ hproj,  // (B, ATTN) f32
    const float* __restrict__ vw,     // (ATTN) f32
    float* __restrict__ scores) {     // (S*B) f32
  __shared__ __bf16 As[2][32 * 32];   // double-buffered 32-row x 32-K bf16 tile
  __shared__ float vw_lds[ATTN_];
  __shared__ float wg_scores[32];

  const int tid  = threadIdx.x;
  const int lane = tid & 31;
  const int wvid = tid >> 5;
  const int mw   = wvid & 1;       // M half: rows mw*16 .. +15 of the 32
  const int nw   = wvid >> 1;      // N quarter: cols nw*256 .. +255
  const int m0   = blockIdx.x * 32;
  const int nbase = nw * 256;

  // stage attn_v_weight into LDS once
  *(float4*)(&vw_lds[tid * 4]) = *(const float4*)(vw + tid * 4);
  if (tid < 32) wg_scores[tid] = 0.f;

  v8f zero = {};
  v8f acc[16];
#pragma unroll
  for (int t = 0; t < 16; ++t) acc[t] = zero;

  // A-tile staging assignment: thread -> (row, 4 consecutive K)
  const int arow = tid >> 3;
  const int akk  = (tid & 7) * 4;
  const float* arow_ptr = v + (size_t)(m0 + arow) * ENC_ + akk;
  // per-lane fragment gather (ISA 16-bit layout):
  //   lanes 0-15:  K {0..7, 16..23};  lanes 16-31: K {8..15, 24..31}
  const int frow = lane & 15;
  const int fkb  = (lane >> 4) * 8;

  // prologue: stage K-step 0
  {
    float4 a4 = *(const float4*)(arow_ptr);
    v4bf a4b;
    a4b[0] = (__bf16)a4.x; a4b[1] = (__bf16)a4.y;
    a4b[2] = (__bf16)a4.z; a4b[3] = (__bf16)a4.w;
    *(v4bf*)(&As[0][arow * 32 + akk]) = a4b;
  }
  __syncthreads();

#pragma unroll 2
  for (int k0 = 0; k0 < KDIM; k0 += 32) {
    const int buf = (k0 >> 5) & 1;
    const bool has_next = (k0 + 32) < KDIM;

    // prefetch the A tile two steps ahead (global_prefetch_b8, no LOADcnt)
    if (k0 + 64 < KDIM) __builtin_prefetch(arow_ptr + k0 + 64, 0, 3);

    // issue next A-tile global load before the WMMAs (latency hidden)
    float4 nx = make_float4(0.f, 0.f, 0.f, 0.f);
    if (has_next) nx = *(const float4*)(arow_ptr + k0 + 32);

    // A fragment for this wave's 16-row sub-tile from current buffer
    v8bf alo = *(const v8bf*)(&As[buf][(mw * 16 + frow) * 32 + fkb]);
    v8bf ahi = *(const v8bf*)(&As[buf][(mw * 16 + frow) * 32 + fkb + 16]);
    v16bf afrag = __builtin_shufflevector(alo, ahi, 0, 1, 2, 3, 4, 5, 6, 7,
                                          8, 9, 10, 11, 12, 13, 14, 15);

#pragma unroll
    for (int t = 0; t < 16; ++t) {
      // B fragment: lane n holds We row (nbase + t*16 + n) over this K chunk
      const __bf16* bp =
          Web + (size_t)(nbase + t * 16 + frow) * ENC_ + k0 + fkb;
      v8bf blo = *(const v8bf*)(bp);
      v8bf bhi = *(const v8bf*)(bp + 16);
      v16bf bfrag = __builtin_shufflevector(blo, bhi, 0, 1, 2, 3, 4, 5, 6, 7,
                                            8, 9, 10, 11, 12, 13, 14, 15);
      acc[t] = __builtin_amdgcn_wmma_f32_16x16x32_bf16(
          false, afrag, false, bfrag, (short)0, acc[t], false, false);
    }

    // convert + store the next tile into the other buffer
    if (has_next) {
      v4bf nb;
      nb[0] = (__bf16)nx.x; nb[1] = (__bf16)nx.y;
      nb[2] = (__bf16)nx.z; nb[3] = (__bf16)nx.w;
      *(v4bf*)(&As[buf ^ 1][arow * 32 + akk]) = nb;
    }
    __syncthreads();
  }

  // epilogue: + hproj, tanh, * vw, reduce over N
  const int hi = lane >> 4;  // C layout: VGPR g holds M=g (lanes 0-15), M=g+8 (16-31)
  int rg[8];
#pragma unroll
  for (int g = 0; g < 8; ++g) rg[g] = m0 + mw * 16 + g + hi * 8;

  float partial[8];
#pragma unroll
  for (int g = 0; g < 8; ++g) partial[g] = 0.f;

#pragma unroll
  for (int t = 0; t < 16; ++t) {
    const int n = nbase + t * 16 + (lane & 15);
    const float vwn = vw_lds[n];
#pragma unroll
    for (int g = 0; g < 8; ++g) {
      float x = acc[t][g] + hproj[(rg[g] & (B_ - 1)) * ATTN_ + n];
      partial[g] += vwn * fast_tanh(x);
    }
  }

#pragma unroll
  for (int g = 0; g < 8; ++g) {
    float p = partial[g];
    p += __shfl_xor(p, 1);
    p += __shfl_xor(p, 2);
    p += __shfl_xor(p, 4);
    p += __shfl_xor(p, 8);
    if ((lane & 15) == 0)
      atomicAdd(&wg_scores[mw * 16 + g + hi * 8], p);  // ds_add_f32
  }
  __syncthreads();
  if (tid < 32) scores[m0 + tid] = wg_scores[tid];
}

// ---------------------------------------------------------------------------
// Kernel 4: per-b softmax over S + context[b,e] = sum_s w[s]*v[s,b,e]
// (attn_v_bias is uniform across s -> softmax-invariant -> dropped)
// ---------------------------------------------------------------------------
__global__ __launch_bounds__(256) void softmax_context_kernel(
    const float* __restrict__ v, const float* __restrict__ scores,
    float* __restrict__ out) {
  __shared__ float wls[S_];
  __shared__ float red[8];
  const int b = blockIdx.x;
  const int tid = threadIdx.x;
  const int lane = tid & 31, wid = tid >> 5;

  float lv[8];
  float lmax = -3.4e38f;
#pragma unroll
  for (int i = 0; i < 8; ++i) {
    float sc = scores[(tid + i * 256) * B_ + b];
    lv[i] = sc;
    lmax = fmaxf(lmax, sc);
  }
  for (int m = 1; m < 32; m <<= 1) lmax = fmaxf(lmax, __shfl_xor(lmax, m));
  if (lane == 0) red[wid] = lmax;
  __syncthreads();
  float bmax = red[0];
#pragma unroll
  for (int i = 1; i < 8; ++i) bmax = fmaxf(bmax, red[i]);

  float lsum = 0.f;
#pragma unroll
  for (int i = 0; i < 8; ++i) {
    float e = __expf(lv[i] - bmax);
    wls[tid + i * 256] = e;
    lsum += e;
  }
  for (int m = 1; m < 32; m <<= 1) lsum += __shfl_xor(lsum, m);
  __syncthreads();
  if (lane == 0) red[wid] = lsum;
  __syncthreads();
  float bsum = 0.f;
#pragma unroll
  for (int i = 0; i < 8; ++i) bsum += red[i];
  const float inv = 1.f / bsum;

  // streaming weighted sum over v[s,b,:]; 256 threads x float4 = 1024 cols
  const int e0 = tid * 4;
  const float* vb = v + (size_t)b * ENC_ + e0;
  float4 acc = make_float4(0.f, 0.f, 0.f, 0.f);
#pragma unroll 4
  for (int s = 0; s < S_; ++s) {
    float ws = wls[s];
    float4 x = *(const float4*)(vb + (size_t)s * (B_ * ENC_));
    acc.x += ws * x.x; acc.y += ws * x.y;
    acc.z += ws * x.z; acc.w += ws * x.w;
  }
  acc.x *= inv; acc.y *= inv; acc.z *= inv; acc.w *= inv;
  *(float4*)(out + (size_t)b * ENC_ + e0) = acc;
}

// ---------------------------------------------------------------------------
extern "C" void kernel_launch(void* const* d_in, const int* in_sizes, int n_in,
                              void* d_out, int out_size, void* d_ws,
                              size_t ws_size, hipStream_t stream) {
  const float* hidden = (const float*)d_in[0];
  const float* v      = (const float*)d_in[1];
  const float* W      = (const float*)d_in[2];
  const float* wb     = (const float*)d_in[3];
  const float* vw     = (const float*)d_in[4];
  // d_in[5] = attn_v_bias: softmax-invariant, unused
  float* out = (float*)d_out;

  char* ws = (char*)d_ws;
  __bf16* Web   = (__bf16*)(ws + WEB_OFF);
  float*  hproj = (float*)(ws + HPROJ_OFF);
  float*  score = (float*)(ws + SCORE_OFF);

  prep_web_kernel<<<(ATTN_ * ENC_ / 4) / 256, 256, 0, stream>>>(W, Web);
  prep_hproj_kernel<<<(B_ * ATTN_) / 256, 256, 0, stream>>>(hidden, W, wb,
                                                            hproj);
  attn_scores_kernel<<<(S_ * B_) / 32, 256, 0, stream>>>(v, Web, hproj, vw,
                                                         score);
  softmax_context_kernel<<<B_, 256, 0, stream>>>(v, score, out);
}
